// VQ_20590073217238
// MI455X (gfx1250) — matrix operand order repaired and checked by reading, not compiled, not run on hardware
//
#include <hip/hip_runtime.h>
#include <hip/hip_bf16.h>
#include <math.h>

// ---------------- problem constants ----------------
#define NN   16384      // B*H*W
#define CB   4          // codebooks
#define DD   64         // dim per codebook
#define KK   1024       // codes
#define ED   256        // CB*DD
#define DECAY 0.99f
#define VEPS  1e-5f

// ---------------- output layout (floats) ----------------
#define QUANT_OFF     0
#define DIFF_OFF      4194304
#define ENC_OFF       4194305
#define PERP_OFF      4259841
#define EMBEDNEW_OFF  4259842
#define CS_OFF        4521986
#define EA_OFF        4526082

// ---------------- workspace layout (bytes) ----------------
#define WS_EMBED_T    0          // 4*1024*64 halves  = 524288 B, [c][k][d] f16
#define WS_E2         524288     // 4096 f32          = 16384 B
#define WS_IND        540672     // 65536 i32         = 262144 B
#define WS_COUNTS     802816     // 4096 f32          = 16384 B
#define WS_ESUM       819200     // 262144 f32        = 1048576 B
#define WS_ACC        1867776    // accums: [0]=diff, [1]=plog; 64 B pad
#define WS_NSUM       1867840    // 4 f32
#define WS_ZERO_FLOATS ((16384 + 1048576 + 128) / 4)   // counts..nsum region

typedef _Float16 v16h __attribute__((ext_vector_type(16)));
typedef float    v8f  __attribute__((ext_vector_type(8)));

// ---------------- zero accumulators ----------------
__global__ void vq_zero(float* p, int n) {
    for (int i = blockIdx.x * blockDim.x + threadIdx.x; i < n; i += gridDim.x * blockDim.x)
        p[i] = 0.0f;
}

// ---------------- precompute: embed^T (f16) + ||e_k||^2 ----------------
__global__ void vq_precompute(const float* __restrict__ embed,
                              _Float16* __restrict__ embed_t,
                              float* __restrict__ e2) {
    int t = blockIdx.x * blockDim.x + threadIdx.x;   // over CB*KK = 4096
    if (t >= CB * KK) return;
    int c = t >> 10;
    int k = t & (KK - 1);
    float s = 0.0f;
    _Float16* row = embed_t + (size_t)t * DD;
    #pragma unroll 4
    for (int d = 0; d < DD; ++d) {
        float v = embed[((size_t)c * DD + d) * KK + k];
        s += v * v;
        row[d] = (_Float16)v;
    }
    e2[t] = s;
}

// ---------------- main: WMMA distance GEMM + argmin ----------------
// grid = (NN/16, CB), block = 128 (4 waves). Wave w handles codes [w*256, w*256+256).
__global__ __launch_bounds__(128)
void vq_argmin(const float* __restrict__ x,
               const _Float16* __restrict__ embed_t,
               const float* __restrict__ e2,
               int* __restrict__ ind) {
    const int n0   = blockIdx.x * 16;
    const int c    = blockIdx.y;
    const int lane = threadIdx.x & 31;
    const int wave = threadIdx.x >> 5;
    const int hi   = lane >> 4;          // lane half
    const int row  = n0 + (lane & 15);   // M index for A / C fragments

    // ---- A fragments (16x32 f16 per fragment): a0 covers d 0..31, a1 covers d 32..63.
    // Layout: lanes 0-15 hold K=0..15 (h0..15); lanes 16-31 hold
    // K=16..31. Convert f32 -> f16 on load; reused across all 16 column tiles.
    const float* frow = x + (size_t)row * ED + c * DD;
    const int b0 = hi ? 16 : 0;
    v16h a0, a1;
    #pragma unroll
    for (int j = 0; j < 8; ++j) {
        a0[j]     = (_Float16)frow[b0 + j];
        a0[j + 8] = (_Float16)frow[b0 + 8 + j];
        a1[j]     = (_Float16)frow[32 + b0 + j];
        a1[j + 8] = (_Float16)frow[32 + b0 + 8 + j];
    }

    float best_d[8];
    int   best_i[8];
    #pragma unroll
    for (int j = 0; j < 8; ++j) { best_d[j] = INFINITY; best_i[j] = 0; }

    // ---- loop over this wave's 16 column tiles (16 codes each)
    for (int t = 0; t < 16; ++t) {
        const int colbase = wave * 256 + t * 16;
        const int col     = colbase + (lane & 15);
        // B fragment: per lane one contiguous run of 16 d-values (32B, aligned)
        const _Float16* brow = embed_t + ((size_t)c * KK + col) * DD;
        v16h bf0 = *(const v16h*)(brow + (hi ? 16 : 0));
        v16h bf1 = *(const v16h*)(brow + 32 + (hi ? 16 : 0));

        v8f acc = {};
        acc = __builtin_amdgcn_wmma_f32_16x16x32_f16(false, a0, false, bf0,
                                                     (short)0, acc, false, false);
        acc = __builtin_amdgcn_wmma_f32_16x16x32_f16(false, a1, false, bf1,
                                                     (short)0, acc, false, false);

        const float ce2 = e2[c * KK + col];
        #pragma unroll
        for (int j = 0; j < 8; ++j) {
            float dist = ce2 - 2.0f * acc[j];      // ||f||^2 constant: irrelevant for argmin
            if (dist < best_d[j]) { best_d[j] = dist; best_i[j] = col; }
        }
    }

    // ---- cross-lane argmin within each 16-lane half (columns live across lanes)
    #pragma unroll
    for (int m = 1; m <= 8; m <<= 1) {
        #pragma unroll
        for (int j = 0; j < 8; ++j) {
            float od = __shfl_xor(best_d[j], m, 32);
            int   oi = __shfl_xor(best_i[j], m, 32);
            if (od < best_d[j] || (od == best_d[j] && oi < best_i[j])) {
                best_d[j] = od; best_i[j] = oi;
            }
        }
    }

    // ---- combine the 4 waves (disjoint, increasing code ranges)
    __shared__ float sd[4][16];
    __shared__ int   si[4][16];
    if ((lane & 15) == 0) {
        #pragma unroll
        for (int j = 0; j < 8; ++j) {
            int r = j + 8 * hi;                    // C layout: VGPR j -> row j (lo) / j+8 (hi)
            sd[wave][r] = best_d[j];
            si[wave][r] = best_i[j];
        }
    }
    __syncthreads();
    if (threadIdx.x < 16) {
        float bd = sd[0][threadIdx.x];
        int   bi = si[0][threadIdx.x];
        #pragma unroll
        for (int w = 1; w < 4; ++w) {
            if (sd[w][threadIdx.x] < bd) { bd = sd[w][threadIdx.x]; bi = si[w][threadIdx.x]; }
        }
        ind[(size_t)(n0 + threadIdx.x) * CB + c] = bi;
    }
}

// ---------------- gather: quantize, diff, embed_sum, counts, enc_ind ----------------
__global__ __launch_bounds__(256)
void vq_gather(const float* __restrict__ x,
               const float* __restrict__ embed,
               const int* __restrict__ ind,
               float* __restrict__ out,
               float* __restrict__ embed_sum,
               float* __restrict__ counts,
               float* __restrict__ accums) {
    __shared__ float red[256];
    const long total = (long)NN * CB * DD;
    float local = 0.0f;
    for (long e = (long)blockIdx.x * blockDim.x + threadIdx.x; e < total;
         e += (long)gridDim.x * blockDim.x) {
        int  d  = (int)(e & 63);
        long nc = e >> 6;                      // n*CB + c
        int  c  = (int)(nc & 3);
        int  k  = ind[nc];
        float qv = embed[((size_t)c * DD + d) * KK + k];
        float fv = x[e];                       // x element order == (n*CB+c)*DD + d
        out[QUANT_OFF + e] = qv;               // f + stopgrad(q - f) == q numerically
        float df = qv - fv;
        local += df * df;
        atomicAdd(&embed_sum[((size_t)c * DD + d) * KK + k], fv);
        if (d == 0) {
            atomicAdd(&counts[c * KK + k], 1.0f);
            out[ENC_OFF + nc] = (float)k;
        }
    }
    red[threadIdx.x] = local;
    __syncthreads();
    for (int s = 128; s > 0; s >>= 1) {
        if (threadIdx.x < s) red[threadIdx.x] += red[threadIdx.x + s];
        __syncthreads();
    }
    if (threadIdx.x == 0) atomicAdd(&accums[0], red[0]);
}

// ---------------- finalize 1: cs_new, per-codebook sums, entropy ----------------
__global__ __launch_bounds__(256)
void vq_finalize1(const float* __restrict__ cluster_size,
                  const float* __restrict__ counts,
                  float* __restrict__ out,
                  float* __restrict__ accums,
                  float* __restrict__ nsums) {
    int c = blockIdx.x;                         // 4 blocks
    __shared__ float s1[256], s2[256];
    float sum_cs = 0.0f, sum_pl = 0.0f;
    for (int k = threadIdx.x; k < KK; k += 256) {
        float cnt = counts[c * KK + k];
        float cs  = DECAY * cluster_size[c * KK + k] + (1.0f - DECAY) * cnt;
        out[CS_OFF + c * KK + k] = cs;
        sum_cs += cs;
        float p = cnt * (1.0f / (float)NN);     // avg_probs
        sum_pl += p * logf(p + 1e-10f);
    }
    s1[threadIdx.x] = sum_cs;
    s2[threadIdx.x] = sum_pl;
    __syncthreads();
    for (int s = 128; s > 0; s >>= 1) {
        if (threadIdx.x < s) {
            s1[threadIdx.x] += s1[threadIdx.x + s];
            s2[threadIdx.x] += s2[threadIdx.x + s];
        }
        __syncthreads();
    }
    if (threadIdx.x == 0) {
        nsums[c] = s1[0];
        atomicAdd(&accums[1], s2[0]);
    }
}

// ---------------- finalize 2: ea_new, embed_new, scalars ----------------
__global__ __launch_bounds__(256)
void vq_finalize2(const float* __restrict__ embed_avg,
                  const float* __restrict__ embed_sum,
                  const float* __restrict__ nsums,
                  const float* __restrict__ accums,
                  float* __restrict__ out) {
    long e = (long)blockIdx.x * 256 + threadIdx.x;   // over CB*DD*KK = 262144
    if (e < (long)CB * DD * KK) {
        int  k  = (int)(e & (KK - 1));
        long cd = e >> 10;                           // c*DD + d
        int  c  = (int)(cd >> 6);
        float ea = DECAY * embed_avg[e] + (1.0f - DECAY) * embed_sum[e];
        out[EA_OFF + e] = ea;
        float cs = out[CS_OFF + c * KK + k];
        float n  = nsums[c];
        float csn = (cs + VEPS) / (n + (float)KK * VEPS) * n;
        out[EMBEDNEW_OFF + e] = ea / csn;
    }
    if (blockIdx.x == 0 && threadIdx.x == 0) {
        out[DIFF_OFF] = accums[0] * (1.0f / (float)(NN * DD));
        out[PERP_OFF] = expf(-accums[1]);
    }
}

// ---------------- launcher ----------------
extern "C" void kernel_launch(void* const* d_in, const int* in_sizes, int n_in,
                              void* d_out, int out_size, void* d_ws, size_t ws_size,
                              hipStream_t stream) {
    const float* x            = (const float*)d_in[0];
    const float* embed        = (const float*)d_in[1];
    const float* cluster_size = (const float*)d_in[2];
    const float* embed_avg    = (const float*)d_in[3];
    float* out = (float*)d_out;
    char*  ws  = (char*)d_ws;

    _Float16* embed_t  = (_Float16*)(ws + WS_EMBED_T);
    float*    e2       = (float*)(ws + WS_E2);
    int*      ind      = (int*)(ws + WS_IND);
    float*    counts   = (float*)(ws + WS_COUNTS);
    float*    esum     = (float*)(ws + WS_ESUM);
    float*    accums   = (float*)(ws + WS_ACC);
    float*    nsums    = (float*)(ws + WS_NSUM);

    // 0) zero the atomic accumulator region (counts + embed_sum + accums/nsums)
    vq_zero<<<1024, 256, 0, stream>>>(counts, WS_ZERO_FLOATS);

    // 1) embed^T in f16 + ||e_k||^2
    vq_precompute<<<(CB * KK + 255) / 256, 256, 0, stream>>>(embed, embed_t, e2);

    // 2) WMMA distance GEMM + argmin
    dim3 grid(NN / 16, CB);
    vq_argmin<<<grid, 128, 0, stream>>>(x, embed_t, e2, ind);

    // 3) gather quantize / diff / scatters
    vq_gather<<<4096, 256, 0, stream>>>(x, embed, ind, out, esum, counts, accums);

    // 4) EMA cluster sizes + entropy
    vq_finalize1<<<CB, 256, 0, stream>>>(cluster_size, counts, out, accums, nsums);

    // 5) EMA embed averages, normalized codebook, scalars
    vq_finalize2<<<(CB * DD * KK + 255) / 256, 256, 0, stream>>>(embed_avg, esum, nsums,
                                                                 accums, out);
}
